// CTNet_42984032698541
// MI455X (gfx1250) — matrix operand order repaired and checked by reading, not compiled, not run on hardware
//
#include <hip/hip_runtime.h>
#include <hip/hip_bf16.h>

typedef _Float16 half_t;
typedef __attribute__((ext_vector_type(16))) _Float16 v16h;
typedef __attribute__((ext_vector_type(8)))  _Float16 v8h;
typedef __attribute__((ext_vector_type(8)))  float    v8f;

static constexpr int BATCH   = 4096;
static constexpr int CHAN    = 512;     // NB_FILTER
static constexpr int CIN     = 7;
static constexpr int S       = 10;      // conv-out spatial
static constexpr int ROWS    = BATCH * S;   // 40960
static constexpr int KIM     = 288;     // im2col K, 280 padded to 288
static constexpr int HIDDEN  = 2048;
static constexpr int FEAT    = 4096;    // 8 * 512 flattened pooled features

// ---------------------------------------------------------------------------
// WMMA GEMM, compile-time N/K:  C[M][N] = A[M][K]*Bt[N][K]^T + bias (opt ReLU)
// A row-major [M][K]; B pre-transposed [N][K] (done once in weight prep) so
// BOTH LDS tiles are K-contiguous: staging is pure b128 load->store, and both
// A and B fragments are contiguous b128 LDS loads using the same contiguous
// k order, k = 16*(lane/16) + s. 256 threads = 8 waves; block tile 256(M) x 64(N); K-step 32;
// wave tile 32x64 = 8 WMMAs per step (2 A frags x 4 shared B frags).
// LDS double-buffered; one barrier per K-step; next tile's global loads are
// issued before this step's WMMAs; K+2 tile prefetched (WGP-scope).
// ---------------------------------------------------------------------------
template <int N, int K, bool RELU, bool OUT16>
__global__ __launch_bounds__(256)
void gemm_wmma(const half_t* __restrict__ A, const half_t* __restrict__ Bt,
               const float* __restrict__ bias,
               half_t* __restrict__ o16, float* __restrict__ o32)
{
    constexpr int KSTEPS = K / 32;
    __shared__ __attribute__((aligned(32))) half_t As[2][256 * 32];
    __shared__ __attribute__((aligned(32))) half_t Bs[2][64 * 32]; // [n][k]

    const int tid  = threadIdx.x;
    const int wave = tid >> 5;
    const int lane = tid & 31;
    const int m0   = blockIdx.x * 256;
    const int n0   = blockIdx.y * 64;

    const int lm   = lane & 15;          // M (A frag) / N (B frag) index
    const int selA = (lane >> 4) << 4;   // A K half: 0 or 16
    const int selB = (lane >> 4) << 4;   // B K half: 0 or 16

    // staging coordinates: A = 512 v16h segs (2/thread), B = 256 v8h segs
    const int ra = tid >> 1, ca = (tid & 1) << 4;
    const int nb = tid >> 2, cb = (tid & 3) << 3;

    auto stageA = [&](int kt, int buf) {
        const int k0 = kt * 32;
        *(v16h*)&As[buf][ra * 32 + ca] =
            *(const v16h*)&A[(size_t)(m0 + ra) * K + k0 + ca];
        *(v16h*)&As[buf][(ra + 128) * 32 + ca] =
            *(const v16h*)&A[(size_t)(m0 + ra + 128) * K + k0 + ca];
    };
    auto stageB = [&](int kt, int buf) {
        const int k0 = kt * 32;
        *(v8h*)&Bs[buf][nb * 32 + cb] =
            *(const v8h*)&Bt[(size_t)(n0 + nb) * K + k0 + cb];
    };

    v8f acc[2][4] = {};
    stageA(0, 0); stageB(0, 0);
    __syncthreads();

    for (int kt = 0; kt < KSTEPS; ++kt) {
        const int cur = kt & 1;
        if (kt + 1 < KSTEPS) { stageA(kt + 1, cur ^ 1); stageB(kt + 1, cur ^ 1); }
        if (kt + 2 < KSTEPS) {
            // locality 3 -> WGP-scope prefetch (warms the near cache)
            __builtin_prefetch(&A[(size_t)(m0 + ra) * K + (kt + 2) * 32 + ca], 0, 3);
            __builtin_prefetch(&Bt[(size_t)(n0 + nb) * K + (kt + 2) * 32 + cb], 0, 3);
        }

        // A fragments: lanes 0-15 K=0..15, lanes 16-31 K=16..31 (contiguous)
        v16h af0, af1;
        {
            const half_t* Ar = &As[cur][(wave * 32 + lm) * 32 + selA];
            *((v8h*)&af0)     = *(const v8h*)(Ar);
            *((v8h*)&af0 + 1) = *(const v8h*)(Ar + 8);
            *((v8h*)&af1)     = *(const v8h*)(Ar + 16 * 32);
            *((v8h*)&af1 + 1) = *(const v8h*)(Ar + 16 * 32 + 8);
        }
#pragma unroll
        for (int nt = 0; nt < 4; ++nt) {
            // B fragment: lanes 0-15 K=0..15, lanes 16-31 K=16..31 (contiguous)
            v16h bf = *(const v16h*)&Bs[cur][(nt * 16 + lm) * 32 + selB];
            acc[0][nt] = __builtin_amdgcn_wmma_f32_16x16x32_f16(
                false, af0, false, bf, (short)0, acc[0][nt], false, false);
            acc[1][nt] = __builtin_amdgcn_wmma_f32_16x16x32_f16(
                false, af1, false, bf, (short)0, acc[1][nt], false, false);
        }
        __syncthreads();
    }

    // epilogue: C/D layout = 8 VGPRs, N = lane&15, M = r + 8*(lane>=16)
#pragma unroll
    for (int sub = 0; sub < 2; ++sub) {
        const int row_base = m0 + wave * 32 + sub * 16 + ((lane >> 4) << 3);
#pragma unroll
        for (int nt = 0; nt < 4; ++nt) {
            const int col = n0 + nt * 16 + lm;
            const float bv = bias[col];
#pragma unroll
            for (int r = 0; r < 8; ++r) {
                float v = acc[sub][nt][r] + bv;
                if (RELU) v = fmaxf(v, 0.f);
                size_t idx = (size_t)(row_base + r) * N + col;
                if (OUT16) o16[idx] = (half_t)v;
                else       o32[idx] = v;
            }
        }
    }
}

// ---------------------------------------------------------------------------
// Prep kernels: f32 -> f16, weights stored pre-transposed as [N][K]
// ---------------------------------------------------------------------------
__global__ void im2col_kernel(const float* __restrict__ x, half_t* __restrict__ P)
{
    size_t idx = (size_t)blockIdx.x * 256 + threadIdx.x; // ROWS*KIM exact
    int kk = (int)(idx % KIM);
    int row = (int)(idx / KIM);
    int s = row % S, b = row / S;
    half_t v = (half_t)0.f;
    if (kk < 280) {
        int ci = kk / 40, rem = kk % 40, kh = rem / 10, kw = rem % 10;
        int w = s - 1 + kw;                      // pad (0,1) in W
        if (w >= 0 && w < 17)
            v = (half_t)x[(((size_t)b * CIN + ci) * 4 + kh) * 17 + w];
    }
    P[idx] = v;
}

// conv weights -> Bt[n][k], k = ci*40 + kh*10 + kw (280 padded to 288)
__global__ void convw_kernel(const float* __restrict__ cw, half_t* __restrict__ W)
{
    int idx = blockIdx.x * 256 + threadIdx.x;    // CHAN*KIM exact
    int n = idx / KIM, k = idx % KIM;
    float v = 0.f;
    if (k < 280) {
        int ci = k / 40, rem = k % 40, kh = rem / 10, kw = rem % 10;
        v = cw[(((size_t)n * CIN + ci) * 4 + kh) * 10 + kw];
    }
    W[idx] = (half_t)v;
}

// q_w/k_w/v_w are [O][C] row-major == [N][K]: pure concat copy-convert
__global__ void qkvw_kernel(const float* __restrict__ qw, const float* __restrict__ kw,
                            const float* __restrict__ vw, half_t* __restrict__ W)
{
    int idx = blockIdx.x * 256 + threadIdx.x;    // 1536*512 exact
    const float* src = (idx < 512 * 512) ? qw : (idx < 2 * 512 * 512) ? kw : vw;
    W[idx] = (half_t)src[idx & (512 * 512 - 1)];
}

__global__ void qkvb_kernel(const float* __restrict__ qb, const float* __restrict__ kb,
                            const float* __restrict__ vb, float* __restrict__ B)
{
    int n = blockIdx.x * 256 + threadIdx.x;      // 1536 exact
    B[n] = (n < 512) ? qb[n] : (n < 1024) ? kb[n - 512] : vb[n - 1024];
}

// fc1_w is [HIDDEN][FEAT] row-major == [N][K]: pure copy-convert
__global__ void fc1w_kernel(const float* __restrict__ fw, half_t* __restrict__ W)
{
    size_t idx = (size_t)blockIdx.x * 256 + threadIdx.x; // HIDDEN*FEAT exact
    W[idx] = (half_t)fw[idx];
}

// ---------------------------------------------------------------------------
// BatchNorm (training mode, biased batch stats over all rows per channel)
// ---------------------------------------------------------------------------
__global__ void zero_stats(float* st) { st[blockIdx.x * 256 + threadIdx.x] = 0.f; }

__global__ __launch_bounds__(256)
void bn_stats_kernel(const half_t* __restrict__ X, float* __restrict__ sum,
                     float* __restrict__ sumsq)
{
    const int t = threadIdx.x;
    const size_t r0 = (size_t)blockIdx.x * 64;
    float s0 = 0.f, q0 = 0.f, s1 = 0.f, q1 = 0.f;
    for (int r = 0; r < 64; ++r) {
        float a = (float)X[(r0 + r) * CHAN + t];
        float b = (float)X[(r0 + r) * CHAN + t + 256];
        s0 += a; q0 += a * a; s1 += b; q1 += b * b;
    }
    atomicAdd(&sum[t],       s0); atomicAdd(&sumsq[t],       q0);
    atomicAdd(&sum[t + 256], s1); atomicAdd(&sumsq[t + 256], q1);
}

__global__ void bn_finalize(const float* __restrict__ sum, const float* __restrict__ sumsq,
                            const float* __restrict__ g, const float* __restrict__ b,
                            float* __restrict__ scale, float* __restrict__ shift,
                            float inv_n)
{
    int c = blockIdx.x * 256 + threadIdx.x;      // 512
    float m   = sum[c] * inv_n;
    float var = sumsq[c] * inv_n - m * m;
    float sc  = g[c] * rsqrtf(var + 1e-5f);
    scale[c] = sc;
    shift[c] = b[c] - m * sc;
}

__global__ void bn_relu_kernel(const half_t* __restrict__ X, const float* __restrict__ scale,
                               const float* __restrict__ shift, half_t* __restrict__ Y)
{
    size_t i = (size_t)blockIdx.x * 256 + threadIdx.x;   // ROWS*CHAN exact
    int c = (int)(i & (CHAN - 1));
    float v = (float)X[i] * scale[c] + shift[c];
    Y[i] = (half_t)fmaxf(v, 0.f);
}

// ---------------------------------------------------------------------------
// MHSA over S=10 positions: one block per batch element.
// qkv layout: row (b*S+s), cols [0,512)=q [512,1024)=k [1024,1536)=v
// ---------------------------------------------------------------------------
__global__ __launch_bounds__(256)
void attn_kernel(const half_t* __restrict__ qkv, const float* __restrict__ rel_h,
                 const float* __restrict__ rel_w, half_t* __restrict__ ao)
{
    __shared__ float qs[S * CHAN];
    __shared__ float ks[S * CHAN];
    __shared__ float ps[S * CHAN];
    __shared__ float logits[S * 12];
    __shared__ float attn[S * 12];

    const int b = blockIdx.x, t = threadIdx.x;
    const size_t base = (size_t)b * S * 1536;

    for (int i = t; i < S * CHAN; i += 256) {
        int s = i >> 9, c = i & (CHAN - 1);
        qs[i] = (float)qkv[base + (size_t)s * 1536 + c];
        ks[i] = (float)qkv[base + (size_t)s * 1536 + 512 + c];
        ps[i] = rel_h[c * S + s] + rel_w[c];     // pos[c][s]
    }
    __syncthreads();

    if (t < S * S) {
        int i = t / S, j = t % S;
        float a = 0.f;
        for (int c = 0; c < CHAN; ++c)
            a += qs[i * CHAN + c] * ks[j * CHAN + c]    // content-content
               + ps[i * CHAN + c] * qs[j * CHAN + c];   // content-position
        logits[i * 12 + j] = a;
    }
    __syncthreads();

    if (t < S) {
        float mx = -1e30f;
        for (int j = 0; j < S; ++j) mx = fmaxf(mx, logits[t * 12 + j]);
        float e[S], sm = 0.f;
        for (int j = 0; j < S; ++j) { e[j] = __expf(logits[t * 12 + j] - mx); sm += e[j]; }
        float inv = 1.f / sm;
        for (int j = 0; j < S; ++j) attn[t * 12 + j] = e[j] * inv;
    }
    __syncthreads();

    for (int i = t; i < S * CHAN; i += 256) {
        int s = i >> 9, c = i & (CHAN - 1);      // out position s, channel c
        float a = 0.f;
#pragma unroll
        for (int j = 0; j < S; ++j)
            a += (float)qkv[base + (size_t)j * 1536 + 1024 + c] * attn[s * 12 + j];
        ao[(size_t)(b * S + s) * CHAN + c] = (half_t)a;
    }
}

// BN2 + ReLU + maxpool(1,3) stride 1 + flatten (c-major: feat = c*8 + w)
__global__ void pool_kernel(const half_t* __restrict__ ao, const float* __restrict__ scale,
                            const float* __restrict__ shift, half_t* __restrict__ pooled)
{
    int idx = blockIdx.x * 256 + threadIdx.x;    // BATCH*CHAN exact
    int c = idx & (CHAN - 1), b = idx >> 9;
    float sc = scale[c], sh = shift[c];
    float v[S];
#pragma unroll
    for (int s = 0; s < S; ++s)
        v[s] = fmaxf((float)ao[(size_t)(b * S + s) * CHAN + c] * sc + sh, 0.f);
#pragma unroll
    for (int w = 0; w < 8; ++w)
        pooled[(size_t)b * FEAT + c * 8 + w] =
            (half_t)fmaxf(fmaxf(v[w], v[w + 1]), v[w + 2]);
}

// FC2 (N=2) + sigmoid: one wave per (batch, class) dot product of 2048
__global__ __launch_bounds__(256)
void fc2_kernel(const float* __restrict__ a1, const float* __restrict__ w,
                const float* __restrict__ bias, float* __restrict__ out)
{
    int wave = threadIdx.x >> 5, lane = threadIdx.x & 31;
    int g = blockIdx.x * 8 + wave;               // 0..8191 = b*2+cls
    int b = g >> 1, cls = g & 1;
    float acc = 0.f;
    for (int h = lane; h < HIDDEN; h += 32)
        acc += a1[(size_t)b * HIDDEN + h] * w[(size_t)cls * HIDDEN + h];
#pragma unroll
    for (int off = 16; off; off >>= 1) acc += __shfl_down(acc, off, 32);
    if (lane == 0) out[g] = 1.f / (1.f + __expf(-(acc + bias[cls])));
}

// ---------------------------------------------------------------------------
extern "C" void kernel_launch(void* const* d_in, const int* in_sizes, int n_in,
                              void* d_out, int out_size, void* d_ws, size_t ws_size,
                              hipStream_t stream)
{
    const float* x      = (const float*)d_in[0];
    const float* conv_w = (const float*)d_in[1];
    const float* conv_b = (const float*)d_in[2];
    const float* bn1_g  = (const float*)d_in[3];
    const float* bn1_b  = (const float*)d_in[4];
    const float* q_w    = (const float*)d_in[5];
    const float* q_b    = (const float*)d_in[6];
    const float* k_w    = (const float*)d_in[7];
    const float* k_b    = (const float*)d_in[8];
    const float* v_w    = (const float*)d_in[9];
    const float* v_b    = (const float*)d_in[10];
    const float* rel_h  = (const float*)d_in[11];
    const float* rel_w  = (const float*)d_in[12];
    const float* bn2_g  = (const float*)d_in[13];
    const float* bn2_b  = (const float*)d_in[14];
    const float* fc1_w  = (const float*)d_in[15];
    const float* fc1_b  = (const float*)d_in[16];
    const float* fc2_w  = (const float*)d_in[17];
    const float* fc2_b  = (const float*)d_in[18];
    float* out = (float*)d_out;

    // workspace layout (256B aligned); buffers reused once their producer's
    // consumers are done (deterministic same-work-every-call)
    char* ws = (char*)d_ws;
    size_t off = 0;
    auto alloc = [&](size_t bytes) { char* p = ws + off; off += (bytes + 255) & ~(size_t)255; return p; };
    half_t* P16    = (half_t*)alloc((size_t)ROWS * KIM * 2);        // im2col patches
    half_t* wA16   = (half_t*)alloc((size_t)CHAN * KIM * 2);        // conv weights [N][K]
    half_t* wqkv16 = (half_t*)alloc((size_t)1536 * CHAN * 2);       // qkv weights [N][K]
    float*  qkvb   = (float*) alloc(1536 * 4);
    half_t* wfc1   = (half_t*)alloc((size_t)HIDDEN * FEAT * 2);     // fc1 weights [N][K]
    half_t* buf1   = (half_t*)alloc((size_t)ROWS * CHAN * 2);       // h_pre, later attn_out
    half_t* buf2   = (half_t*)alloc((size_t)ROWS * CHAN * 2);       // xf,   later pooled
    half_t* buf3   = (half_t*)alloc((size_t)ROWS * 1536 * 2);       // qkv,  later fc1 acts (f32)
    float*  stats  = (float*) alloc(8 * CHAN * 4);                  // sums/scales x2 BNs
    half_t* hpre   = buf1;          half_t* ao     = buf1;
    half_t* xf     = buf2;          half_t* pooled = buf2;
    half_t* qkv    = buf3;          float*  a1     = (float*)buf3;
    float* sum1 = stats, *sq1 = stats + 512, *sc1 = stats + 1024, *sh1 = stats + 1536;
    float* sum2 = stats + 2048, *sq2 = stats + 2560, *sc2 = stats + 3072, *sh2 = stats + 3584;
    const float inv_n = 1.f / (float)ROWS;

    // weight prep + im2col
    convw_kernel <<<  576, 256, 0, stream>>>(conv_w, wA16);
    qkvw_kernel  <<< 3072, 256, 0, stream>>>(q_w, k_w, v_w, wqkv16);
    qkvb_kernel  <<<    6, 256, 0, stream>>>(q_b, k_b, v_b, qkvb);
    fc1w_kernel  <<<32768, 256, 0, stream>>>(fc1_w, wfc1);
    im2col_kernel<<<46080, 256, 0, stream>>>(x, P16);
    zero_stats   <<<   16, 256, 0, stream>>>(stats);

    // conv as GEMM: [40960 x 288] x [288 x 512] + conv_b -> h_pre
    gemm_wmma<CHAN, KIM, false, true><<<dim3(ROWS / 256, CHAN / 64), 256, 0, stream>>>(
        P16, wA16, conv_b, hpre, nullptr);

    // BN1 (batch stats) + ReLU
    bn_stats_kernel<<<ROWS / 64, 256, 0, stream>>>(hpre, sum1, sq1);
    bn_finalize    <<<2, 256, 0, stream>>>(sum1, sq1, bn1_g, bn1_b, sc1, sh1, inv_n);
    bn_relu_kernel <<<(ROWS * CHAN) / 256, 256, 0, stream>>>(hpre, sc1, sh1, xf);

    // fused QKV GEMM: [40960 x 512] x [512 x 1536] + bias
    gemm_wmma<1536, CHAN, false, true><<<dim3(ROWS / 256, 1536 / 64), 256, 0, stream>>>(
        xf, wqkv16, qkvb, qkv, nullptr);

    // attention (S=10) per batch element
    attn_kernel<<<BATCH, 256, 0, stream>>>(qkv, rel_h, rel_w, ao);

    // BN2 + ReLU + maxpool + flatten
    bn_stats_kernel<<<ROWS / 64, 256, 0, stream>>>(ao, sum2, sq2);
    bn_finalize    <<<2, 256, 0, stream>>>(sum2, sq2, bn2_g, bn2_b, sc2, sh2, inv_n);
    pool_kernel    <<<(BATCH * CHAN) / 256, 256, 0, stream>>>(ao, sc2, sh2, pooled);

    // FC1 GEMM + ReLU: [4096 x 4096] x [4096 x 2048] -> f32 activations
    gemm_wmma<HIDDEN, FEAT, true, false><<<dim3(BATCH / 256, HIDDEN / 64), 256, 0, stream>>>(
        pooled, wfc1, fc1_b, nullptr, a1);

    // FC2 + sigmoid
    fc2_kernel<<<1024, 256, 0, stream>>>(a1, fc2_w, fc2_b, out);
}
